// TF_Sparse_Cosine_Attention_5755256177024
// MI455X (gfx1250) — matrix operand order repaired and checked by reading, not compiled, not run on hardware
//
#include <hip/hip_runtime.h>
#include <math.h>

// ---------------------------------------------------------------------------
// TF Sparse Cosine Attention, MI455X (gfx1250, wave32, WMMA bf16)
// B=32768, LATENT=256, DK=64, TF=1536, HID=128
// ---------------------------------------------------------------------------

#define Bn     32768
#define LATENT 256
#define DK     64
#define TF     1536
#define HID    128
#define SROW   (TF + 4)   // padded LDS row stride (dwords): kills bank conflicts

typedef __bf16 v16bf __attribute__((ext_vector_type(16)));
typedef float  v8f   __attribute__((ext_vector_type(8)));

__device__ __forceinline__ v8f wmma_bf16(v16bf a, v16bf b, v8f c) {
    // D = A(16x32 bf16) * B(32x16 bf16) + C(16x16 f32)
    return __builtin_amdgcn_wmma_f32_16x16x32_bf16(
        /*neg_a=*/false, a, /*neg_b=*/false, b,
        /*c_mod=*/(short)0, c, /*reuse_a=*/false, /*reuse_b=*/false);
}

// Build a 16-bit A-fragment (16x32 tile, this wave's K-step) from an fp32 row.
// Lane l holds row M=l%16; khalf=l/16 selects K-subsets {0..15} / {16..31}.
__device__ __forceinline__ v16bf load_a_frag_f32(const float* rowptr, int k0, int khalf) {
    const float4* pa = reinterpret_cast<const float4*>(rowptr + k0 + khalf * 16);
    const float4* pb = reinterpret_cast<const float4*>(rowptr + k0 + khalf * 16 + 8);
    float4 a0 = pa[0], a1 = pa[1];
    float4 b0 = pb[0], b1 = pb[1];
    v16bf r;
    r[0]  = (__bf16)a0.x; r[1]  = (__bf16)a0.y; r[2]  = (__bf16)a0.z; r[3]  = (__bf16)a0.w;
    r[4]  = (__bf16)a1.x; r[5]  = (__bf16)a1.y; r[6]  = (__bf16)a1.z; r[7]  = (__bf16)a1.w;
    r[8]  = (__bf16)b0.x; r[9]  = (__bf16)b0.y; r[10] = (__bf16)b0.z; r[11] = (__bf16)b0.w;
    r[12] = (__bf16)b1.x; r[13] = (__bf16)b1.y; r[14] = (__bf16)b1.z; r[15] = (__bf16)b1.w;
    return r;
}

// A-fragment permutation for a 32-wide K group: storage order
// [g+0..31] (identity) -> contiguous v16bf loads per lane-half.
__device__ __forceinline__ int aperm_k(int i) {
    int g = (i >> 5) << 5;
    int w = i & 31;
    int k = w;
    return g + k;
}

// ---------------------------------------------------------------------------
// Kernel 1: transpose + bf16-convert weights into [n][k] (K contiguous) layout
// ---------------------------------------------------------------------------
__global__ void k_prep_weights(const float* __restrict__ WQ,
                               const float* __restrict__ W1,
                               const float* __restrict__ W2,
                               __bf16* __restrict__ WQt,   // [64][256]
                               __bf16* __restrict__ W1t,   // [128][1536]
                               __bf16* __restrict__ W2t) { // [1536][128]
    const int N0 = DK * LATENT;       // 16384
    const int N1 = HID * TF;          // 196608
    const int N2 = TF * HID;          // 196608
    int i = blockIdx.x * blockDim.x + threadIdx.x;
    int stride = gridDim.x * blockDim.x;
    for (; i < N0 + N1 + N2; i += stride) {
        if (i < N0) {
            int n = i / LATENT, k = i % LATENT;
            WQt[i] = (__bf16)WQ[k * DK + n];
        } else if (i < N0 + N1) {
            int j = i - N0;
            int n = j / TF, k = j % TF;
            W1t[j] = (__bf16)W1[k * HID + n];
        } else {
            int j = i - N0 - N1;
            int n = j / HID, k = j % HID;
            W2t[j] = (__bf16)W2[k * TF + n];
        }
    }
}

// ---------------------------------------------------------------------------
// Kernel 2: Kn = normalize_rows(E_TF @ W_K) / tau  -> bf16 [TF][DK]
// one block (64 threads) per TF row
// ---------------------------------------------------------------------------
__global__ __launch_bounds__(64) void k_prep_kn(const float* __restrict__ E_TF,
                                                const float* __restrict__ W_K,
                                                const float* __restrict__ tau_p,
                                                __bf16* __restrict__ Knp) {
    __shared__ float Erow[LATENT];
    __shared__ float red[DK + 1];
    int t = blockIdx.x;
    int n = threadIdx.x;                 // 0..63
    for (int j = n; j < LATENT; j += 64) Erow[j] = E_TF[t * LATENT + j];
    __syncthreads();
    float acc = 0.0f;
    for (int k = 0; k < LATENT; ++k) acc = fmaf(Erow[k], W_K[k * DK + n], acc);
    red[n] = acc * acc;
    __syncthreads();
    if (n == 0) {
        float s = 0.0f;
        for (int j = 0; j < DK; ++j) s += red[j];
        red[DK] = sqrtf(s);
    }
    __syncthreads();
    float inv = 1.0f / ((red[DK] + 1e-8f) * tau_p[0]);   // fold 1/tau into Kn
    Knp[t * DK + n] = (__bf16)(acc * inv);
}

// ---------------------------------------------------------------------------
// Kernel 3: Qn = normalize_rows(z @ W_Q), stored bf16 in A-fragment-permuted
// layout [B][64].  1 wave -> 16x64 tile, K=256 (8 WMMA k-steps x 4 n-tiles).
// ---------------------------------------------------------------------------
__global__ __launch_bounds__(256) void k_prep_qn(const float* __restrict__ z,
                                                 const __bf16* __restrict__ WQt,
                                                 __bf16* __restrict__ Qnp) {
    __shared__ float stage[8][16][DK];   // 32 KB
    const int lane  = threadIdx.x & 31;
    const int wave  = threadIdx.x >> 5;
    const int m     = lane & 15;
    const int khalf = lane >> 4;
    const int rowBase = blockIdx.x * 128 + wave * 16;

    const float* zrow = z + (size_t)(rowBase + m) * LATENT;
    v8f acc[4] = {};
    for (int s = 0; s < 8; ++s) {
        v16bf a = load_a_frag_f32(zrow, s * 32, khalf);
        #pragma unroll
        for (int nt = 0; nt < 4; ++nt) {
            v16bf b = *reinterpret_cast<const v16bf*>(
                WQt + (nt * 16 + m) * LATENT + s * 32 + khalf * 16);
            acc[nt] = wmma_bf16(a, b, acc[nt]);
        }
    }
    #pragma unroll
    for (int nt = 0; nt < 4; ++nt)
        #pragma unroll
        for (int r = 0; r < 8; ++r)
            stage[wave][r + 8 * khalf][nt * 16 + m] = acc[nt][r];
    __syncthreads();

    if (lane < 16) {
        float ss = 0.0f;
        for (int j = 0; j < DK; ++j) { float v = stage[wave][lane][j]; ss = fmaf(v, v, ss); }
        float inv = 1.0f / (sqrtf(ss) + 1e-8f);
        __bf16* out = Qnp + (size_t)(rowBase + lane) * DK;
        for (int i = 0; i < DK; ++i)
            out[i] = (__bf16)(stage[wave][lane][aperm_k(i)] * inv);
    }
}

// ---------------------------------------------------------------------------
// Kernel 4: H = relu(motif @ W1 + b1), stored bf16 A-fragment-permuted [B][128]
// 1 wave -> 16x128 tile, K=1536 (48 WMMA k-steps x 8 n-tiles).
// ---------------------------------------------------------------------------
__global__ __launch_bounds__(256) void k_prep_hidden(const float* __restrict__ motif,
                                                     const __bf16* __restrict__ W1t,
                                                     const float* __restrict__ b1,
                                                     __bf16* __restrict__ Hnp) {
    __shared__ __bf16 stage[8][16][HID]; // 32 KB
    const int lane  = threadIdx.x & 31;
    const int wave  = threadIdx.x >> 5;
    const int m     = lane & 15;
    const int khalf = lane >> 4;
    const int rowBase = blockIdx.x * 128 + wave * 16;

    const float* mrow = motif + (size_t)(rowBase + m) * TF;
    v8f acc[8] = {};
    for (int s = 0; s < 48; ++s) {
        v16bf a = load_a_frag_f32(mrow, s * 32, khalf);
        #pragma unroll
        for (int nt = 0; nt < 8; ++nt) {
            v16bf b = *reinterpret_cast<const v16bf*>(
                W1t + (nt * 16 + m) * TF + s * 32 + khalf * 16);
            acc[nt] = wmma_bf16(a, b, acc[nt]);
        }
    }
    #pragma unroll
    for (int nt = 0; nt < 8; ++nt) {
        float bias = b1[nt * 16 + m];
        #pragma unroll
        for (int r = 0; r < 8; ++r) {
            float v = fmaxf(acc[nt][r] + bias, 0.0f);
            stage[wave][r + 8 * khalf][nt * 16 + m] = (__bf16)v;
        }
    }
    __syncthreads();

    if (lane < 16) {
        __bf16* out = Hnp + (size_t)(rowBase + lane) * HID;
        for (int i = 0; i < HID; ++i)
            out[i] = stage[wave][lane][aperm_k(i)];
    }
}

// ---------------------------------------------------------------------------
// Kernel 5 (fused): S = Qn@Kn^T (1/tau folded) + H@W2 + b2, then row-wise
// sparsemax via Michelot threshold iteration.
// 4 waves cooperate on one 16-row block: each wave computes 24 of the 96
// column tiles into the shared padded f32 score block (96.25 KB LDS), then
// each wave runs sparsemax on 4 rows out of VGPRs.
// 3 blocks/WGP (LDS-bound) -> 12 waves/WGP = 3 per SIMD32.
// ---------------------------------------------------------------------------
__global__ __launch_bounds__(128) void k_fused(const __bf16* __restrict__ Qnp,
                                               const __bf16* __restrict__ Hnp,
                                               const __bf16* __restrict__ Knp,
                                               const __bf16* __restrict__ W2t,
                                               const float*  __restrict__ b2,
                                               float* __restrict__ out) {
    extern __shared__ float S[];                 // [16][SROW]
    const int lane  = threadIdx.x & 31;
    const int wave  = threadIdx.x >> 5;          // 0..3
    const int m     = lane & 15;
    const int khalf = lane >> 4;
    const int rowBase = blockIdx.x * 16;

    // A fragments for this 16-row block (pre-permuted -> contiguous loads);
    // all 4 waves load the same rows (cheap, L0/L2 resident).
    const __bf16* q = Qnp + (size_t)(rowBase + m) * DK;
    v16bf aQ0 = *reinterpret_cast<const v16bf*>(q + khalf * 16);
    v16bf aQ1 = *reinterpret_cast<const v16bf*>(q + 32 + khalf * 16);
    const __bf16* h = Hnp + (size_t)(rowBase + m) * HID;
    v16bf aH[4];
    #pragma unroll
    for (int s = 0; s < 4; ++s)
        aH[s] = *reinterpret_cast<const v16bf*>(h + s * 32 + khalf * 16);

    // each wave owns 24 of the 96 column tiles
    const int ntEnd = wave * 24 + 24;
    for (int nt = wave * 24; nt < ntEnd; ++nt) {
        v8f acc = {};
        const __bf16* kb = Knp + (nt * 16 + m) * DK + khalf * 16;
        acc = wmma_bf16(aQ0, *reinterpret_cast<const v16bf*>(kb),      acc);
        acc = wmma_bf16(aQ1, *reinterpret_cast<const v16bf*>(kb + 32), acc);
        const __bf16* wb = W2t + (nt * 16 + m) * HID + khalf * 16;
        #pragma unroll
        for (int s = 0; s < 4; ++s)
            acc = wmma_bf16(aH[s], *reinterpret_cast<const v16bf*>(wb + s * 32), acc);
        float bias = b2[nt * 16 + m];
        #pragma unroll
        for (int r = 0; r < 8; ++r)              // C layout: M=r+8*khalf, N=m
            S[(r + 8 * khalf) * SROW + nt * 16 + m] = acc[r] + bias;
    }
    __syncthreads();

    // Row-wise sparsemax (sort-free): tau <- (sum_{s>tau} s - 1)/|{s>tau}|,
    // initialized with tau0 = (sum - 1)/TF; converges finitely & monotonically.
    // Each wave handles 4 of the 16 rows; 48 elements per lane in VGPRs.
    const int rEnd = wave * 4 + 4;
    for (int r = wave * 4; r < rEnd; ++r) {
        float v[TF / 32];
        const float* Sr = S + r * SROW;
        float tsum = 0.0f;
        #pragma unroll
        for (int j = 0; j < TF / 32; ++j) { v[j] = Sr[lane + j * 32]; tsum += v[j]; }
        #pragma unroll
        for (int off = 16; off; off >>= 1) tsum += __shfl_xor(tsum, off, 32);
        float tauv = (tsum - 1.0f) * (1.0f / (float)TF);
        for (int it = 0; it < 8; ++it) {
            float cs = 0.0f, cn = 0.0f;
            #pragma unroll
            for (int j = 0; j < TF / 32; ++j)
                if (v[j] > tauv) { cs += v[j]; cn += 1.0f; }
            #pragma unroll
            for (int off = 16; off; off >>= 1) {
                cs += __shfl_xor(cs, off, 32);
                cn += __shfl_xor(cn, off, 32);
            }
            tauv = (cs - 1.0f) / fmaxf(cn, 1.0f);
        }
        float* orow = out + (size_t)(rowBase + r) * TF;
        #pragma unroll
        for (int j = 0; j < TF / 32; ++j)
            orow[lane + j * 32] = fmaxf(v[j] - tauv, 0.0f);
    }
}

// ---------------------------------------------------------------------------
extern "C" void kernel_launch(void* const* d_in, const int* in_sizes, int n_in,
                              void* d_out, int out_size, void* d_ws, size_t ws_size,
                              hipStream_t stream) {
    const float* z     = (const float*)d_in[0];
    const float* motif = (const float*)d_in[1];
    const float* E_TF  = (const float*)d_in[2];
    const float* W_Q   = (const float*)d_in[3];
    const float* W_K   = (const float*)d_in[4];
    const float* tau   = (const float*)d_in[5];
    const float* W1    = (const float*)d_in[6];
    const float* b1    = (const float*)d_in[7];
    const float* W2    = (const float*)d_in[8];
    const float* b2    = (const float*)d_in[9];
    float* out = (float*)d_out;

    // workspace carve-up (256B aligned), total ~13.6 MB
    char* ws = (char*)d_ws;
    size_t off = 0;
    auto carve = [&](size_t bytes) -> void* {
        void* p = ws + off;
        off = (off + bytes + 255) & ~(size_t)255;
        return p;
    };
    __bf16* WQt = (__bf16*)carve((size_t)DK * LATENT * 2);   // [64][256]
    __bf16* W1t = (__bf16*)carve((size_t)HID * TF * 2);      // [128][1536]
    __bf16* W2t = (__bf16*)carve((size_t)TF * HID * 2);      // [1536][128]
    __bf16* Knp = (__bf16*)carve((size_t)TF * DK * 2);       // [1536][64]
    __bf16* Qnp = (__bf16*)carve((size_t)Bn * DK * 2);       // [B][64]
    __bf16* Hnp = (__bf16*)carve((size_t)Bn * HID * 2);      // [B][128]

    k_prep_weights<<<512, 256, 0, stream>>>(W_Q, W1, W2, WQt, W1t, W2t);
    k_prep_kn<<<TF, 64, 0, stream>>>(E_TF, W_K, tau, Knp);
    k_prep_qn<<<Bn / 128, 256, 0, stream>>>(z, WQt, Qnp);
    k_prep_hidden<<<Bn / 128, 256, 0, stream>>>(motif, W1t, b1, Hnp);
    k_fused<<<Bn / 16, 128, 16 * SROW * sizeof(float), stream>>>(Qnp, Hnp, Knp, W2t, b2, out);
}
